// RegionalRouter_59064390255199
// MI455X (gfx1250) — matrix operand order repaired and checked by reading, not compiled, not run on hardware
//
#include <hip/hip_runtime.h>
#include <stdint.h>

// Problem constants (from reference): B=4, N=40000, D=512, H=128, E=16, K=2
#define BN_TOKENS 160000
#define DIM_D 512
#define DIM_H 128
#define DIM_E 16
#define TOPK 2

#define TM 256          // tokens per block (8 waves x 32 tokens)
#define TPW 32          // tokens per wave (two 16-row A tiles per B fragment)
#define NWAVES 8
#define W1T_LD 520      // padded row stride (elements) for W1^T in LDS (bank-conflict break)

typedef __bf16 bf16_t;
typedef bf16_t v16bf __attribute__((ext_vector_type(16)));
typedef float  v8f   __attribute__((ext_vector_type(8)));

union FragAB { v16bf v; uint32_t u[8]; };
union FragC  { v8f   v; float    f[8]; };

// round-to-nearest-even fp32 -> bf16, packed pair into one dword
__device__ __forceinline__ uint32_t pack2_bf16(float a, float b) {
    uint32_t ua = __float_as_uint(a);
    uint32_t ub = __float_as_uint(b);
    ua += 0x7FFFu + ((ua >> 16) & 1u);
    ub += 0x7FFFu + ((ub >> 16) & 1u);
    return (ua >> 16) | (ub & 0xFFFF0000u);
}

__device__ __forceinline__ uint16_t cvt_bf16(float a) {
    uint32_t ua = __float_as_uint(a);
    ua += 0x7FFFu + ((ua >> 16) & 1u);
    return (uint16_t)(ua >> 16);
}

// 16 contiguous bf16 (32B) from LDS into a fragment (two ds_load_b128)
__device__ __forceinline__ void load_frag_lds(FragAB& f, const uint16_t* p) {
    uint4 w0 = *(const uint4*)(p);
    uint4 w1 = *(const uint4*)(p + 8);
    f.u[0] = w0.x; f.u[1] = w0.y; f.u[2] = w0.z; f.u[3] = w0.w;
    f.u[4] = w1.x; f.u[5] = w1.y; f.u[6] = w1.z; f.u[7] = w1.w;
}

// A-fragment (16-bit layout): chunks [abase..abase+7] and [abase+16..abase+23]
__device__ __forceinline__ void load_afrag_lds(FragAB& f, const uint16_t* p) {
    uint4 w0 = *(const uint4*)(p);
    uint4 w1 = *(const uint4*)(p + 16);
    f.u[0] = w0.x; f.u[1] = w0.y; f.u[2] = w0.z; f.u[3] = w0.w;
    f.u[4] = w1.x; f.u[5] = w1.y; f.u[6] = w1.z; f.u[7] = w1.w;
}

__device__ __forceinline__ void pack_afrag(FragAB& a, const float4& f0, const float4& f1,
                                           const float4& f2, const float4& f3) {
    a.u[0] = pack2_bf16(f0.x, f0.y);
    a.u[1] = pack2_bf16(f0.z, f0.w);
    a.u[2] = pack2_bf16(f1.x, f1.y);
    a.u[3] = pack2_bf16(f1.z, f1.w);
    a.u[4] = pack2_bf16(f2.x, f2.y);
    a.u[5] = pack2_bf16(f2.z, f2.w);
    a.u[6] = pack2_bf16(f3.x, f3.y);
    a.u[7] = pack2_bf16(f3.z, f3.w);
}

__global__ void __launch_bounds__(256, 1)
regional_router_kernel(const float* __restrict__ x,
                       const float* __restrict__ node_regions,
                       const float* __restrict__ W1,
                       const float* __restrict__ b1,
                       const float* __restrict__ W2,
                       const float* __restrict__ b2,
                       const float* __restrict__ rbias,
                       float* __restrict__ gates_out,
                       int*   __restrict__ experts_out)
{
    extern __shared__ char smem_raw[];
    // LDS carve-up:
    //   W1^T bf16 [DIM_H][W1T_LD]            : 128*520*2  = 133120 B
    //   W2^T bf16 [DIM_E][DIM_H]             : 16*128*2   =   4096 B
    //   h tiles  bf16 [NWAVES][TPW][DIM_H]   : 8*32*128*2 =  65536 B
    //   logits   f32  [NWAVES][TPW][DIM_E]   : 8*32*16*4  =  16384 B
    uint16_t* W1t = (uint16_t*)smem_raw;
    uint16_t* W2t = (uint16_t*)(smem_raw + DIM_H * W1T_LD * 2);
    uint16_t* hs  = (uint16_t*)(smem_raw + DIM_H * W1T_LD * 2 + DIM_E * DIM_H * 2);
    float*    lgs = (float*)   (smem_raw + DIM_H * W1T_LD * 2 + DIM_E * DIM_H * 2
                                         + NWAVES * TPW * DIM_H * 2);

    const int tid  = threadIdx.x;
    const int wave = tid >> 5;
    const int lane = tid & 31;

    // ---------------- Stage weights (bf16, transposed) into LDS ----------------
    // W1[k][h] (row-major, h fastest) -> W1t[h][k]
    for (int i = tid; i < DIM_D * DIM_H; i += 256) {
        int k = i >> 7;       // 0..511
        int h = i & 127;      // 0..127
        W1t[h * W1T_LD + k] = cvt_bf16(W1[i]);
    }
    // W2[h][e] -> W2t[e][h]
    for (int i = tid; i < DIM_H * DIM_E; i += 256) {
        int h = i >> 4;
        int e = i & 15;
        W2t[e * DIM_H + h] = cvt_bf16(W2[i]);
    }
    __syncthreads();

    const int tokBase = blockIdx.x * TM + wave * TPW;
    const int m     = lane & 15;          // row (A/C) or column (B) within 16
    const int hi    = lane >> 4;          // lane half
    const int abase = hi ? 8 : 0;         // A-fragment K sub-base (16-bit layout)
    const int bbase = hi ? 16 : 0;        // unused; B fragments also load at abase
    // Two token tiles per wave: rows m and m+16 of the 32-token slab.
    const float* xb0 = x + (long long)(tokBase + m)      * DIM_D + abase;
    const float* xb1 = x + (long long)(tokBase + 16 + m) * DIM_D + abase;

    // ---------------- GEMM1: h = relu(x_slab[32xD] @ W1[DxH] + b1) ----------------
    FragC acc[2][8];
#pragma unroll
    for (int t = 0; t < 2; ++t)
#pragma unroll
        for (int nt = 0; nt < 8; ++nt)
#pragma unroll
            for (int j = 0; j < 8; ++j) acc[t][nt].f[j] = 0.0f;

    // Software pipeline: A-source floats for K-step 0 in flight before the loop.
    float4 p00 = *(const float4*)(xb0);      float4 p01 = *(const float4*)(xb0 + 4);
    float4 p02 = *(const float4*)(xb0 + 16); float4 p03 = *(const float4*)(xb0 + 20);
    float4 p10 = *(const float4*)(xb1);      float4 p11 = *(const float4*)(xb1 + 4);
    float4 p12 = *(const float4*)(xb1 + 16); float4 p13 = *(const float4*)(xb1 + 20);

    for (int ks = 0; ks < DIM_D / 32; ++ks) {   // 16 K-steps of 32
        FragAB a0, a1;
        pack_afrag(a0, p00, p01, p02, p03);
        pack_afrag(a1, p10, p11, p12, p13);

        // Prefetch A-source floats for next K-step (overlaps with WMMA chain).
        if (ks < DIM_D / 32 - 1) {
            const float* q0 = xb0 + (ks + 1) * 32;
            const float* q1 = xb1 + (ks + 1) * 32;
            p00 = *(const float4*)(q0);      p01 = *(const float4*)(q0 + 4);
            p02 = *(const float4*)(q0 + 16); p03 = *(const float4*)(q0 + 20);
            p10 = *(const float4*)(q1);      p11 = *(const float4*)(q1 + 4);
            p12 = *(const float4*)(q1 + 16); p13 = *(const float4*)(q1 + 20);
        }

        // Each B fragment feeds TWO wmmas (token tiles 0 and 1): halves the
        // exposed LDS latency per wmma and halves DS traffic.
#pragma unroll
        for (int nt = 0; nt < 8; ++nt) {
            FragAB bb;
            load_afrag_lds(bb, W1t + (nt * 16 + m) * W1T_LD + ks * 32 + abase);
            acc[0][nt].v = __builtin_amdgcn_wmma_f32_16x16x32_bf16(
                false, a0.v, false, bb.v, (short)0, acc[0][nt].v, false, false);
            acc[1][nt].v = __builtin_amdgcn_wmma_f32_16x16x32_bf16(
                false, a1.v, false, bb.v, (short)0, acc[1][nt].v, false, false);
        }
    }

    // Epilogue GEMM1: +b1, ReLU, bf16 -> LDS h tile [TPW][DIM_H]
    uint16_t* hw = hs + wave * TPW * DIM_H;
#pragma unroll
    for (int nt = 0; nt < 8; ++nt) {
        int hcol = nt * 16 + m;         // this lane's C column
        float bias = b1[hcol];
#pragma unroll
        for (int t = 0; t < 2; ++t) {
#pragma unroll
            for (int r = 0; r < 8; ++r) {
                int row = t * 16 + r + (hi ? 8 : 0);   // C layout: VGPR r -> M=r / M=r+8
                float v = acc[t][nt].f[r] + bias;
                v = v > 0.0f ? v : 0.0f;
                hw[row * DIM_H + hcol] = cvt_bf16(v);
            }
        }
    }

    // ---------------- GEMM2: logits = h[32x128] @ W2[128x16] ----------------
    FragC acc2[2];
#pragma unroll
    for (int t = 0; t < 2; ++t)
#pragma unroll
        for (int j = 0; j < 8; ++j) acc2[t].f[j] = 0.0f;

    // W2 fragments shared by both token tiles.
    FragAB bw[4];
#pragma unroll
    for (int ks = 0; ks < DIM_H / 32; ++ks)
        load_afrag_lds(bw[ks], W2t + m * DIM_H + ks * 32 + abase);

#pragma unroll
    for (int ks = 0; ks < DIM_H / 32; ++ks) {
        FragAB a20, a21;
        load_afrag_lds(a20, hw + m * DIM_H        + ks * 32 + abase);
        load_afrag_lds(a21, hw + (16 + m) * DIM_H + ks * 32 + abase);
        acc2[0].v = __builtin_amdgcn_wmma_f32_16x16x32_bf16(
            false, a20.v, false, bw[ks].v, (short)0, acc2[0].v, false, false);
        acc2[1].v = __builtin_amdgcn_wmma_f32_16x16x32_bf16(
            false, a21.v, false, bw[ks].v, (short)0, acc2[1].v, false, false);
    }

    // Scatter raw logits to LDS [token][expert]
    float* lw = lgs + wave * TPW * DIM_E;
#pragma unroll
    for (int t = 0; t < 2; ++t)
#pragma unroll
        for (int r = 0; r < 8; ++r) {
            int row = t * 16 + r + (hi ? 8 : 0);
            lw[row * DIM_E + m] = acc2[t].f[r];
        }

    // Same-wave LDS write->read ordering
    asm volatile("s_wait_dscnt 0" ::: "memory");

    // ---------------- Per-token top-2 + softmax (all 32 lanes, one token each) ----------------
    {
        int token = tokBase + lane;
        if (token < BN_TOKENS) {
            const float* nr = node_regions + (long long)token * DIM_E;
            float4 nr0 = *(const float4*)(nr);
            float4 nr1 = *(const float4*)(nr + 4);
            float4 nr2 = *(const float4*)(nr + 8);
            float4 nr3 = *(const float4*)(nr + 12);
            float nrv[16] = { nr0.x, nr0.y, nr0.z, nr0.w,
                              nr1.x, nr1.y, nr1.z, nr1.w,
                              nr2.x, nr2.y, nr2.z, nr2.w,
                              nr3.x, nr3.y, nr3.z, nr3.w };
            float best0 = -3.4e38f, best1 = -3.4e38f;
            int   i0 = 0, i1 = 0;
#pragma unroll
            for (int e = 0; e < DIM_E; ++e) {
                float v = lw[lane * DIM_E + e] + b2[e] + rbias[e] * nrv[e];
                if (v > best0) { best1 = best0; i1 = i0; best0 = v; i0 = e; }
                else if (v > best1) { best1 = v; i1 = e; }
            }
            // softmax over [best0, best1] (descending, matches top_k order)
            float e1 = __expf(best1 - best0);
            float inv = 1.0f / (1.0f + e1);
            gates_out[token * TOPK + 0]   = inv;
            gates_out[token * TOPK + 1]   = e1 * inv;
            experts_out[token * TOPK + 0] = i0;
            experts_out[token * TOPK + 1] = i1;
        }
    }
}

extern "C" void kernel_launch(void* const* d_in, const int* in_sizes, int n_in,
                              void* d_out, int out_size, void* d_ws, size_t ws_size,
                              hipStream_t stream) {
    const float* x   = (const float*)d_in[0];
    const float* nr  = (const float*)d_in[1];
    const float* W1  = (const float*)d_in[2];
    const float* b1  = (const float*)d_in[3];
    const float* W2  = (const float*)d_in[4];
    const float* b2  = (const float*)d_in[5];
    const float* rb  = (const float*)d_in[6];

    // d_out = [gates (B*N*K floats) | selected_experts (B*N*K ints)]
    float* gates   = (float*)d_out;
    int*   experts = (int*)((float*)d_out + (size_t)BN_TOKENS * TOPK);

    size_t shmem = (size_t)DIM_H * W1T_LD * 2        // W1^T bf16
                 + (size_t)DIM_E * DIM_H * 2         // W2^T bf16
                 + (size_t)NWAVES * TPW * DIM_H * 2  // h tiles bf16
                 + (size_t)NWAVES * TPW * DIM_E * 4; // logits f32

    hipFuncSetAttribute((const void*)regional_router_kernel,
                        hipFuncAttributeMaxDynamicSharedMemorySize, (int)shmem);

    int blocks = (BN_TOKENS + TM - 1) / TM;  // 625
    regional_router_kernel<<<blocks, 256, shmem, stream>>>(
        x, nr, W1, b1, W2, b2, rb, gates, experts);
}
